// SpatialAttention_73169062855034
// MI455X (gfx1250) — matrix operand order repaired and checked by reading, not compiled, not run on hardware
//
#include <hip/hip_runtime.h>
#include <hip/hip_bf16.h>
#include <math.h>

// ---------------------------------------------------------------------------
// SpatialAttention (CBAM spatial gate), MI455X / gfx1250, wave32.
//   x:[32,112,112,128] f32  ->  out = x * sigmoid(conv7x7(concat(mean_c, max_c)))
// Phase 1: channel mean/max  (memory bound, reads 205MB)
// Phase 2: 7x7 conv + sigmoid as im2col GEMM on v_wmma_f32_16x16x32_f16
// Phase 3: broadcast gating multiply (memory bound, 410MB)
// ---------------------------------------------------------------------------

#define BB_ 32
#define HH_ 112
#define WW_ 112
#define CC_ 128
#define NPIX (BB_ * HH_ * WW_)      // 401408
#define KREAL 98                    // 7*7*2 patch length
#define KPAD 128                    // padded K, 4 chunks of 32

typedef __attribute__((ext_vector_type(16))) _Float16 v16h;
typedef __attribute__((ext_vector_type(8)))  float    v8f;

// ---------------------------------------------------------------------------
// Kernel 1: per-pixel channel mean + max. One wave (32 lanes) per pixel,
// each lane loads one float4 (32*4 = 128 channels), wave32 xor-shuffle reduce.
// ---------------------------------------------------------------------------
__global__ void sa_stats_kernel(const float* __restrict__ x,
                                float2* __restrict__ stats)
{
    const int wid  = (int)((blockIdx.x * blockDim.x + threadIdx.x) >> 5); // pixel
    const int lane = (int)(threadIdx.x & 31);

    const float4 v = ((const float4*)(x + (size_t)wid * CC_))[lane];

    float s = v.x + v.y + v.z + v.w;
    float m = fmaxf(fmaxf(v.x, v.y), fmaxf(v.z, v.w));

    #pragma unroll
    for (int off = 16; off >= 1; off >>= 1) {
        s += __shfl_xor(s, off, 32);
        m  = fmaxf(m, __shfl_xor(m, off, 32));
    }

    if (lane == 0)
        stats[wid] = make_float2(s * (1.0f / (float)CC_), m);
}

// ---------------------------------------------------------------------------
// Kernel 2: 7x7 conv (2->1 ch, zero 'SAME' padding) + bias + sigmoid via WMMA.
// One wave computes 16 consecutive output pixels (flat pixel order).
//   D(16x16) += A(16x32 f16) * B(32x16 f16), 4 K-chunks (K padded 98->128).
// A: lane l holds pixel m=l&15; slot s holds patch element K = s+8*((s>>3)+g),
//    g = l>>4 (per ISA 16-bit A 16x32 VGPR layout).
// B: lane l holds column n=l&15; slot s = same K row as A slot s, same tap in every column
//    (filter replicated across columns so every D column equals the conv).
// D: VGPR j, lane group g -> pixel j + 8*g (all 16 lanes of a group identical;
//    lanes 0 and 16 write).
// ---------------------------------------------------------------------------
__global__ void sa_conv_wmma_kernel(const float2* __restrict__ stats,
                                    const float* __restrict__ Wf,
                                    const float* __restrict__ bias,
                                    float* __restrict__ gate)
{
    const int wid  = (int)((blockIdx.x * blockDim.x + threadIdx.x) >> 5); // tile
    const int lane = (int)(threadIdx.x & 31);
    const int base = wid * 16;

    const int m = lane & 15;          // pixel within tile held by this lane (A)
    const int g = lane >> 4;          // lane group (K half selector)

    // decode this lane's pixel coordinates
    const int p   = base + m;
    const int b   = p / (HH_ * WW_);
    const int rem = p - b * (HH_ * WW_);
    const int y   = rem / WW_;
    const int xq  = rem - y * WW_;

    const float bval = bias[0];

    v8f acc = {};

    #pragma unroll
    for (int ck = 0; ck < 4; ++ck) {
        // ---- build A fragment (patch gather, zero-padded borders / K tail)
        v16h a;
        #pragma unroll
        for (int s = 0; s < 16; ++s) {
            const int K  = s + 8 * ((s >> 3) + g);
            const int kk = ck * 32 + K;
            float v = 0.0f;
            if (kk < KREAL) {
                const int dy = kk / 14;
                const int r2 = kk - dy * 14;
                const int dx = r2 >> 1;
                const int c  = r2 & 1;
                const int yy = y  + dy - 3;
                const int xx = xq + dx - 3;
                if (yy >= 0 && yy < HH_ && xx >= 0 && xx < WW_) {
                    const float* sp =
                        (const float*)&stats[((size_t)b * HH_ + yy) * WW_ + xx];
                    v = sp[c];
                }
            }
            a[s] = (_Float16)v;
        }

        // ---- build B fragment: slot s = K row s + 8*((s>>3)+g), same tap for all N
        v16h bm;
        #pragma unroll
        for (int s = 0; s < 16; ++s) {
            const int kkb = ck * 32 + s + 8 * ((s >> 3) + g);
            bm[s] = (kkb < KREAL) ? (_Float16)Wf[kkb] : (_Float16)0.0f;
        }

        // D = A*B + C   (f32 accumulate)
        acc = __builtin_amdgcn_wmma_f32_16x16x32_f16(
            /*neg_a=*/false, a, /*neg_b=*/false, bm,
            /*c_mod=*/(short)0, acc, /*reuse_a=*/false, /*reuse_b=*/false);
    }

    // ---- extract: VGPR j of lane-group g holds pixel base + j + 8*g.
    // All 16 lanes of a group hold identical values; one lane per group writes.
    if ((lane & 15) == 0) {
        #pragma unroll
        for (int j = 0; j < 8; ++j) {
            const float attn = acc[j] + bval;
            gate[base + 8 * g + j] = 1.0f / (1.0f + __expf(-attn));
        }
    }
}

// ---------------------------------------------------------------------------
// Kernel 3: out = x * gate (gate broadcast over 128 channels).
// One float4 per thread; 32 consecutive threads share one gate value.
// ---------------------------------------------------------------------------
__global__ void sa_apply_kernel(const float* __restrict__ x,
                                const float* __restrict__ gate,
                                float* __restrict__ out)
{
    const size_t i = (size_t)blockIdx.x * blockDim.x + threadIdx.x; // float4 idx
    const float gv = gate[i >> 5];                 // 128 floats = 32 float4 / pixel
    const float4 v = ((const float4*)x)[i];
    ((float4*)out)[i] = make_float4(v.x * gv, v.y * gv, v.z * gv, v.w * gv);
}

// ---------------------------------------------------------------------------
extern "C" void kernel_launch(void* const* d_in, const int* in_sizes, int n_in,
                              void* d_out, int out_size, void* d_ws, size_t ws_size,
                              hipStream_t stream)
{
    const float* x  = (const float*)d_in[0];   // [32,112,112,128] f32
    const float* Wf = (const float*)d_in[1];   // [7,7,2,1] f32 (HWIO, flat == our kk order)
    const float* bf = (const float*)d_in[2];   // [1] f32
    float* out      = (float*)d_out;

    // scratch: stats float2[NPIX] then gate float[NPIX]
    float2* stats = (float2*)d_ws;
    float*  gate  = (float*)((char*)d_ws + (size_t)NPIX * sizeof(float2));

    // Phase 1: 1 wave / pixel, 8 waves / block -> 50176 blocks
    sa_stats_kernel<<<NPIX / 8, 256, 0, stream>>>(x, stats);

    // Phase 2: 1 wave / 16 pixels -> 25088 waves -> 3136 blocks of 8 waves
    sa_conv_wmma_kernel<<<(NPIX / 16) / 8, 256, 0, stream>>>(stats, Wf, bf, gate);

    // Phase 3: 1 float4 / thread -> 12845056 threads -> 50176 blocks
    sa_apply_kernel<<<((size_t)NPIX * CC_ / 4) / 256, 256, 0, stream>>>(x, gate, out);
}
